// GAT_78589311582293
// MI455X (gfx1250) — compile-verified, matrix operand order checked by reading, not run on hardware
//
#include <hip/hip_runtime.h>
#include <hip/hip_bf16.h>

// ---------------------------------------------------------------------------
// GAT (2x GATConv + linear skips) for MI455X / gfx1250.
// Dense projections: v_wmma_f32_16x16x32_f16 (f16 in, f32 accumulate),
// A-tiles staged global->LDS via async-to-LDS (ASYNCcnt) when available.
// Edge softmax-aggregation: ordered-uint atomicMax + f32 atomicAdd scatter.
// ---------------------------------------------------------------------------

typedef __attribute__((ext_vector_type(16))) _Float16 v16h;
typedef __attribute__((ext_vector_type(8)))  float    v8f;

// GCC-style vector to match the async-to-LDS builtin's parameter type
typedef int gat_v4i __attribute__((vector_size(16)));
typedef __attribute__((address_space(1))) gat_v4i* gat_gptr;
typedef __attribute__((address_space(3))) gat_v4i* gat_lptr;

union Frag16 { uint4 q[2]; v16h h; };

#if __has_builtin(__builtin_amdgcn_global_load_async_to_lds_b128)
#define GAT_HAS_ASYNC_LDS 1
#else
#define GAT_HAS_ASYNC_LDS 0
#endif

// ---- ordered-float encoding for atomic max on f32 --------------------------
__device__ __forceinline__ unsigned fenc(float f) {
    unsigned u = __float_as_uint(f);
    return (u & 0x80000000u) ? ~u : (u | 0x80000000u);
}
__device__ __forceinline__ float fdec(unsigned k) {
    unsigned u = (k & 0x80000000u) ? (k & 0x7FFFFFFFu) : ~k;
    return __uint_as_float(u);
}
__device__ __forceinline__ float leaky(float v) { return v > 0.f ? v : 0.2f * v; }

// ---------------------------------------------------------------------------
// Utility kernels
// ---------------------------------------------------------------------------
__global__ void k_fill_u32(unsigned* __restrict__ p, unsigned v, long count) {
    long i = (long)blockIdx.x * blockDim.x + threadIdx.x;
    if (i < count) p[i] = v;
}

// cast fp32 [rows,128] -> f16 [rowsPad,128], zeroing pad rows
__global__ void k_cast_pad(const float* __restrict__ src, _Float16* __restrict__ dst,
                           int rows, long total) {
    long i = (long)blockIdx.x * blockDim.x + threadIdx.x;
    if (i >= total) return;
    long r = i >> 7;
    dst[i] = (r < rows) ? (_Float16)src[i] : (_Float16)0.f;
}

// Bt1[256][128] = fused [W1 | Wl1] transposed, f16
__global__ void k_build_bt1(const float* __restrict__ W1, const float* __restrict__ Wl1,
                            _Float16* __restrict__ Bt) {
    int i = blockIdx.x * blockDim.x + threadIdx.x;
    if (i >= 256 * 128) return;
    int n = i >> 7, k = i & 127;
    float v = (n < 128) ? W1[k * 128 + n] : Wl1[k * 128 + (n - 128)];
    Bt[i] = (_Float16)v;
}

// Bt2[320][128] = fused [W3 (240) | Wl3 (40) | zero-pad (40)] transposed, f16
__global__ void k_build_bt2(const float* __restrict__ W3, const float* __restrict__ Wl3,
                            _Float16* __restrict__ Bt) {
    int i = blockIdx.x * blockDim.x + threadIdx.x;
    if (i >= 320 * 128) return;
    int n = i >> 7, k = i & 127;
    float v = 0.f;
    if (n < 240)       v = W3[k * 240 + n];
    else if (n < 280)  v = Wl3[k * 40 + (n - 240)];
    Bt[i] = (_Float16)v;
}

// ---------------------------------------------------------------------------
// WMMA GEMM: C[Mpad, NPAD] = A[Mpad,128] (f16, row-major) x Bt[NPAD,128]^T
// block = 256 threads = 8 waves; block tile 64x64.
// Each wave owns two 16x16 C tiles stacked in M (shared B fragment).
// A block tile (64x128 halves = 16KB) staged in LDS (async-to-LDS if present).
// ---------------------------------------------------------------------------
template <int NPAD>
__global__ __launch_bounds__(256)
void k_gemm_wmma(const _Float16* __restrict__ A, const _Float16* __restrict__ Bt,
                 float* __restrict__ C) {
    __shared__ __align__(16) _Float16 As[64 * 128];  // 16 KB
    const int tid  = threadIdx.x;
    const int wave = tid >> 5;
    const int lane = tid & 31;
    const int m_blk = blockIdx.x * 64;
    const int n_blk = blockIdx.y * 64;
    const int wr = wave & 1;   // 0..1 : which 32-row half
    const int wc = wave >> 1;  // 0..3 : which 16-col strip

    // ---- stage A[64 x 128] into LDS: 1024 x b128 --------------------------
#if GAT_HAS_ASYNC_LDS
    {
        const char* gsrc = (const char*)(A + (size_t)m_blk * 128);
#pragma unroll
        for (int t = 0; t < 4; t++) {
            int off = (tid + t * 256) * 16;
            __builtin_amdgcn_global_load_async_to_lds_b128(
                (gat_gptr)(gsrc + off),
                (gat_lptr)((char*)As + off),
                0, 0);
        }
#if __has_builtin(__builtin_amdgcn_s_wait_asynccnt)
        __builtin_amdgcn_s_wait_asynccnt(0);
#else
        asm volatile("s_wait_asynccnt 0x0" ::: "memory");
#endif
    }
#else
    {
        const uint4* Ag  = reinterpret_cast<const uint4*>(A + (size_t)m_blk * 128);
        uint4*       As4 = reinterpret_cast<uint4*>(As);
#pragma unroll
        for (int t = 0; t < 4; t++) As4[tid + t * 256] = Ag[tid + t * 256];
    }
#endif
    __syncthreads();

    // fragment addressing (16-bit A 16x32 layout; B mirrored via transposed Bt)
    const int row = lane & 15;
    const int kb  = (lane >> 4) << 3;  // 0 or 8
    const _Float16* ap0 = As + (size_t)(wr * 32 + row) * 128;
    const _Float16* ap1 = ap0 + 16 * 128;
    const _Float16* bp  = Bt + (size_t)(n_blk + wc * 16 + row) * 128;

    v8f acc0 = {}, acc1 = {};
#pragma unroll
    for (int k0 = 0; k0 < 128; k0 += 32) {
        Frag16 a0, a1, b;
        b.q[0]  = *reinterpret_cast<const uint4*>(bp + k0 + kb);
        b.q[1]  = *reinterpret_cast<const uint4*>(bp + k0 + 16 + kb);
        a0.q[0] = *reinterpret_cast<const uint4*>(ap0 + k0 + kb);
        a0.q[1] = *reinterpret_cast<const uint4*>(ap0 + k0 + 16 + kb);
        a1.q[0] = *reinterpret_cast<const uint4*>(ap1 + k0 + kb);
        a1.q[1] = *reinterpret_cast<const uint4*>(ap1 + k0 + 16 + kb);
        acc0 = __builtin_amdgcn_wmma_f32_16x16x32_f16(false, a0.h, false, b.h,
                                                      (short)0, acc0, false, false);
        acc1 = __builtin_amdgcn_wmma_f32_16x16x32_f16(false, a1.h, false, b.h,
                                                      (short)0, acc1, false, false);
    }

    // C/D 16x16 f32 layout: vgpr i, lanes 0-15 -> M=i, lanes 16-31 -> M=8+i
    const int crow = (lane >> 4) << 3;
    const int ccol = lane & 15;
    float* cp0 = C + (size_t)(m_blk + wr * 32 + crow) * NPAD + n_blk + wc * 16 + ccol;
    float* cp1 = cp0 + (size_t)16 * NPAD;
#pragma unroll
    for (int i = 0; i < 8; i++) {
        cp0[i * NPAD] = acc0[i];   // compile-time strides -> immediate offsets
        cp1[i * NPAD] = acc1[i];
    }
}

// ---------------------------------------------------------------------------
// per-(node, head) attention coefficients:  a_s = <h[n,h,:], a_src[h,:]>
// ---------------------------------------------------------------------------
__global__ void k_alpha(const float* __restrict__ C, const float* __restrict__ asrc,
                        const float* __restrict__ adst, float* __restrict__ as_,
                        float* __restrict__ ad_, int n_nodes, int H, int Cw, int stride) {
    int idx = blockIdx.x * blockDim.x + threadIdx.x;
    if (idx >= n_nodes * H) return;
    int n = idx / H, h = idx - n * H;
    const float* hp = C + (size_t)n * stride + h * Cw;
    const float* ws = asrc + h * Cw;
    const float* wd = adst + h * Cw;
    float s = 0.f, d = 0.f;
    for (int c = 0; c < Cw; c++) { float v = hp[c]; s += v * ws[c]; d += v * wd[c]; }
    as_[idx] = s;
    ad_[idx] = d;
}

// ---------------------------------------------------------------------------
// edge passes: segment-max, exp-sum, message scatter
// ---------------------------------------------------------------------------
__global__ void k_edge_max(const int* __restrict__ ei, int E, int Etot,
                           const float* __restrict__ as_, const float* __restrict__ ad_,
                           unsigned* __restrict__ mk, int H) {
    int e = blockIdx.x * blockDim.x + threadIdx.x;
    if (e >= Etot) return;
    int src, dst;
    if (e < E) { src = ei[e]; dst = ei[E + e]; } else { src = dst = e - E; }
    for (int h = 0; h < H; h++) {
        float lg = leaky(as_[src * H + h] + ad_[dst * H + h]);
        atomicMax(&mk[dst * H + h], fenc(lg));
    }
}

__global__ void k_edge_sum(const int* __restrict__ ei, int E, int Etot,
                           const float* __restrict__ as_, const float* __restrict__ ad_,
                           const unsigned* __restrict__ mk, float* __restrict__ denom, int H) {
    int e = blockIdx.x * blockDim.x + threadIdx.x;
    if (e >= Etot) return;
    int src, dst;
    if (e < E) { src = ei[e]; dst = ei[E + e]; } else { src = dst = e - E; }
    for (int h = 0; h < H; h++) {
        float lg = leaky(as_[src * H + h] + ad_[dst * H + h]);
        atomicAdd(&denom[dst * H + h], __expf(lg - fdec(mk[dst * H + h])));
    }
}

// layer 1: H=4, 32 ch/head, h-features at C1[src*256 + h*32 + c]; one wave/edge
__global__ void k_edge_msg1(const int* __restrict__ ei, int E, int Etot,
                            const float* __restrict__ as_, const float* __restrict__ ad_,
                            const unsigned* __restrict__ mk, const float* __restrict__ denom,
                            const float* __restrict__ C1, float* __restrict__ gat) {
    int wid  = (int)(((long)blockIdx.x * blockDim.x + threadIdx.x) >> 5);
    int lane = threadIdx.x & 31;
    if (wid >= Etot) return;
    int src, dst;
    if (wid < E) { src = ei[wid]; dst = ei[E + wid]; } else { src = dst = wid - E; }
    float al[4];
#pragma unroll
    for (int h = 0; h < 4; h++) {
        float lg = leaky(as_[src * 4 + h] + ad_[dst * 4 + h]);
        float ee = __expf(lg - fdec(mk[dst * 4 + h]));
        al[h] = ee / (denom[dst * 4 + h] + 1e-16f);
    }
    const float* hs = C1 + (size_t)src * 256;
    float*       go = gat + (size_t)dst * 128;
#pragma unroll
    for (int h = 0; h < 4; h++)
        atomicAdd(&go[h * 32 + lane], hs[h * 32 + lane] * al[h]);
}

// layer 2: H=6, 40 ch/head (240 total), features at C2[src*320 + c]; one wave/edge
__global__ void k_edge_msg2(const int* __restrict__ ei, int E, int Etot,
                            const float* __restrict__ as_, const float* __restrict__ ad_,
                            const unsigned* __restrict__ mk, const float* __restrict__ denom,
                            const float* __restrict__ C2, float* __restrict__ gat) {
    int wid  = (int)(((long)blockIdx.x * blockDim.x + threadIdx.x) >> 5);
    int lane = threadIdx.x & 31;
    if (wid >= Etot) return;
    int src, dst;
    if (wid < E) { src = ei[wid]; dst = ei[E + wid]; } else { src = dst = wid - E; }
    float al[6];
#pragma unroll
    for (int h = 0; h < 6; h++) {
        float lg = leaky(as_[src * 6 + h] + ad_[dst * 6 + h]);
        float ee = __expf(lg - fdec(mk[dst * 6 + h]));
        al[h] = ee / (denom[dst * 6 + h] + 1e-16f);
    }
    const float* hs = C2 + (size_t)src * 320;
    float*       go = gat + (size_t)dst * 240;
#pragma unroll
    for (int j = 0; j < 8; j++) {
        int c = lane + 32 * j;
        if (c < 240) atomicAdd(&go[c], hs[c] * al[c / 40]);
    }
}

// ---------------------------------------------------------------------------
// epilogues
// ---------------------------------------------------------------------------
// h = elu(gat + b1 + xWl1 + bl1); write f16 for layer-2 GEMM
__global__ void k_finalize1(const float* __restrict__ C1, const float* __restrict__ gat,
                            const float* __restrict__ b1, const float* __restrict__ bl1,
                            _Float16* __restrict__ h16, int n_nodes) {
    long i = (long)blockIdx.x * blockDim.x + threadIdx.x;
    if (i >= (long)n_nodes * 128) return;
    int j = (int)(i & 127);
    float g = gat[i] + b1[j] + C1[(i >> 7) * 256 + 128 + j] + bl1[j];
    float e = g > 0.f ? g : (__expf(g) - 1.f);
    h16[i] = (_Float16)e;
}

// out = log_softmax(mean_heads(gat2) + b3 + hWl3 + bl3); one wave per node
__global__ void k_finalize2(const float* __restrict__ C2, const float* __restrict__ gat,
                            const float* __restrict__ b3, const float* __restrict__ bl3,
                            float* __restrict__ out, int n_nodes) {
    int wid  = (int)(((long)blockIdx.x * blockDim.x + threadIdx.x) >> 5);
    int lane = threadIdx.x & 31;
    if (wid >= n_nodes) return;
    const float* g   = gat + (size_t)wid * 240;
    const float* lin = C2 + (size_t)wid * 320 + 240;

    float v0, v1;
    {
        int c = lane;  // 0..31 (< 40 always valid)
        float s = 0.f;
#pragma unroll
        for (int h = 0; h < 6; h++) s += g[h * 40 + c];
        v0 = s * (1.f / 6.f) + b3[c] + lin[c] + bl3[c];
    }
    if (lane < 8) {
        int c = lane + 32;
        float s = 0.f;
#pragma unroll
        for (int h = 0; h < 6; h++) s += g[h * 40 + c];
        v1 = s * (1.f / 6.f) + b3[c] + lin[c] + bl3[c];
    } else {
        v1 = -3.4e38f;
    }
    float m = fmaxf(v0, v1);
#pragma unroll
    for (int off = 16; off; off >>= 1) m = fmaxf(m, __shfl_xor(m, off, 32));
    float s = __expf(v0 - m) + ((lane < 8) ? __expf(v1 - m) : 0.f);
#pragma unroll
    for (int off = 16; off; off >>= 1) s += __shfl_xor(s, off, 32);
    float ls = __logf(s);
    float* o = out + (size_t)wid * 40;
    o[lane] = v0 - m - ls;
    if (lane < 8) o[lane + 32] = v1 - m - ls;
}

// ---------------------------------------------------------------------------
extern "C" void kernel_launch(void* const* d_in, const int* in_sizes, int n_in,
                              void* d_out, int out_size, void* d_ws, size_t ws_size,
                              hipStream_t stream) {
    const float* x      = (const float*)d_in[0];
    const int*   ei     = (const int*)d_in[1];   // int32 (JAX default x64-off)
    const float* W1     = (const float*)d_in[2];
    const float* a_src1 = (const float*)d_in[3];
    const float* a_dst1 = (const float*)d_in[4];
    const float* b1     = (const float*)d_in[5];
    const float* Wl1    = (const float*)d_in[6];
    const float* bl1    = (const float*)d_in[7];
    const float* W3     = (const float*)d_in[8];
    const float* a_src3 = (const float*)d_in[9];
    const float* a_dst3 = (const float*)d_in[10];
    const float* b3     = (const float*)d_in[11];
    const float* Wl3    = (const float*)d_in[12];
    const float* bl3    = (const float*)d_in[13];
    float* out = (float*)d_out;

    const int Nn   = in_sizes[0] / 128;
    const int E    = in_sizes[1] / 2;
    const int Etot = E + Nn;
    const int Mpad = ((Nn + 63) / 64) * 64;   // 64-row block tiles

    // ---- bump allocator over workspace (buffers reused across layers) ----
    char* p = (char*)d_ws;
    auto alloc = [&](size_t bytes) {
        void* r = (void*)p;
        p += (bytes + 255) & ~(size_t)255;
        return r;
    };
    _Float16* Af16 = (_Float16*)alloc((size_t)Mpad * 128 * 2);  // x_f16, then h_f16
    _Float16* Bt   = (_Float16*)alloc((size_t)320 * 128 * 2);   // fused W^T, per layer
    float*    C    = (float*)alloc((size_t)Mpad * 320 * 4);     // GEMM out, per layer
    float*    as_  = (float*)alloc((size_t)Nn * 6 * 4);
    float*    ad_  = (float*)alloc((size_t)Nn * 6 * 4);
    unsigned* mk   = (unsigned*)alloc((size_t)Nn * 6 * 4);
    float*    den  = (float*)alloc((size_t)Nn * 6 * 4);
    float*    gat  = (float*)alloc((size_t)Nn * 240 * 4);

    const int TB = 256;
    auto blocks = [&](long n) { return (unsigned)((n + TB - 1) / TB); };
    const int eb = (int)blocks(Etot);

    // ================= layer 1 =================
    {
        long tot = (long)Mpad * 128;
        k_cast_pad<<<blocks(tot), TB, 0, stream>>>(x, Af16, Nn, tot);
        k_build_bt1<<<blocks(256 * 128), TB, 0, stream>>>(W1, Wl1, Bt);
        dim3 g1(Mpad / 64, 256 / 64);
        k_gemm_wmma<256><<<g1, TB, 0, stream>>>(Af16, Bt, C);

        k_alpha<<<blocks((long)Nn * 4), TB, 0, stream>>>(C, a_src1, a_dst1, as_, ad_,
                                                         Nn, 4, 32, 256);
        k_fill_u32<<<blocks((long)Nn * 4), TB, 0, stream>>>(mk, 0u, (long)Nn * 4);
        k_fill_u32<<<blocks((long)Nn * 4), TB, 0, stream>>>((unsigned*)den, 0u, (long)Nn * 4);
        k_fill_u32<<<blocks((long)Nn * 128), TB, 0, stream>>>((unsigned*)gat, 0u,
                                                              (long)Nn * 128);
        k_edge_max<<<eb, TB, 0, stream>>>(ei, E, Etot, as_, ad_, mk, 4);
        k_edge_sum<<<eb, TB, 0, stream>>>(ei, E, Etot, as_, ad_, mk, den, 4);
        k_edge_msg1<<<(Etot + 7) / 8, TB, 0, stream>>>(ei, E, Etot, as_, ad_, mk, den, C, gat);
        k_finalize1<<<blocks((long)Nn * 128), TB, 0, stream>>>(C, gat, b1, bl1, Af16, Nn);
    }

    // ================= layer 2 =================
    {
        k_build_bt2<<<blocks(320 * 128), TB, 0, stream>>>(W3, Wl3, Bt);
        dim3 g2(Mpad / 64, 320 / 64);
        k_gemm_wmma<320><<<g2, TB, 0, stream>>>(Af16, Bt, C);

        k_alpha<<<blocks((long)Nn * 6), TB, 0, stream>>>(C, a_src3, a_dst3, as_, ad_,
                                                         Nn, 6, 40, 320);
        k_fill_u32<<<blocks((long)Nn * 6), TB, 0, stream>>>(mk, 0u, (long)Nn * 6);
        k_fill_u32<<<blocks((long)Nn * 6), TB, 0, stream>>>((unsigned*)den, 0u, (long)Nn * 6);
        k_fill_u32<<<blocks((long)Nn * 240), TB, 0, stream>>>((unsigned*)gat, 0u,
                                                              (long)Nn * 240);
        k_edge_max<<<eb, TB, 0, stream>>>(ei, E, Etot, as_, ad_, mk, 6);
        k_edge_sum<<<eb, TB, 0, stream>>>(ei, E, Etot, as_, ad_, mk, den, 6);
        k_edge_msg2<<<(Etot + 7) / 8, TB, 0, stream>>>(ei, E, Etot, as_, ad_, mk, den, C, gat);
        k_finalize2<<<(Nn + 7) / 8, TB, 0, stream>>>(C, gat, b3, bl3, out, Nn);
    }

    (void)n_in; (void)out_size; (void)ws_size;
}
